// DynaConvBlock_26774826124045
// MI455X (gfx1250) — matrix operand order repaired and checked by reading, not compiled, not run on hardware
//
#include <hip/hip_runtime.h>

// ---------------------------------------------------------------------------
// CDNA5 (gfx1250, wave32) WMMA types
// ---------------------------------------------------------------------------
typedef __attribute__((ext_vector_type(16))) __bf16 v16bf;
typedef __attribute__((ext_vector_type(8)))  __bf16 v8bf;
typedef __attribute__((ext_vector_type(8)))  float  v8f;
typedef __attribute__((ext_vector_type(4)))  int    v4i;

__device__ __forceinline__ v8f wmma_bf16(v16bf a, v16bf b, v8f c) {
    // (neg_a, A, neg_b, B, c_mod, C, reuse_a, reuse_b)
    return __builtin_amdgcn_wmma_f32_16x16x32_bf16(false, a, false, b,
                                                   (short)0, c, false, false);
}

// A-fragment (16x32 bf16): lane l<16 -> row l, K = kk+hi*16 .. +15,
// klo = kk + hi*8.  Two contiguous 16B chunks per lane.
__device__ __forceinline__ v16bf lds_frag_a(const __bf16* rowp, int klo) {
    union { v16bf v; v8bf h[2]; } u;
    const int o = klo + ((threadIdx.x & 31) >> 4) * 8;
    u.h[0] = *(const v8bf*)(rowp + o);
    u.h[1] = *(const v8bf*)(rowp + o + 8);
    return u.v;
}
// B-fragment (32x16 bf16), K-contiguous per column: lane l<16 -> col l,
// K = kk+hi*16 .. +15 (one 32B contiguous run).
__device__ __forceinline__ v16bf lds_frag_b(const __bf16* p) {
    union { v16bf v; v8bf h[2]; } u;
    u.h[0] = *(const v8bf*)(p);
    u.h[1] = *(const v8bf*)(p + 8);
    return u.v;
}
// Global f32 -> bf16 fragments
__device__ __forceinline__ v16bf gld_cvt_a(const float* rowp, int klo) {
    v16bf r;
    const int o = klo + ((threadIdx.x & 31) >> 4) * 8;
#pragma unroll
    for (int i = 0; i < 8; ++i) r[i] = (__bf16)rowp[o + i];
#pragma unroll
    for (int i = 0; i < 8; ++i) r[8 + i] = (__bf16)rowp[o + 8 + i];
    return r;
}
__device__ __forceinline__ v16bf gld_cvt_b(const float* p) {
    v16bf r;
#pragma unroll
    for (int i = 0; i < 16; ++i) r[i] = (__bf16)p[i];
    return r;
}

// ---------------------------------------------------------------------------
// Async global->LDS (ASYNCcnt) path, guarded so compilation never regresses
// ---------------------------------------------------------------------------
#if defined(__has_builtin)
#if __has_builtin(__builtin_amdgcn_global_load_async_to_lds_b128)
#define USE_ASYNC_LDS 1
#endif
#endif
#ifndef USE_ASYNC_LDS
#define USE_ASYNC_LDS 0
#endif

#if USE_ASYNC_LDS
__device__ __forceinline__ void async_copy16(const void* g, void* l) {
    __builtin_amdgcn_global_load_async_to_lds_b128(
        (__attribute__((address_space(1))) v4i*)g,
        (__attribute__((address_space(3))) v4i*)l, 0, 0);
}
#if __has_builtin(__builtin_amdgcn_s_wait_asynccnt)
#define WAIT_ASYNC(n) __builtin_amdgcn_s_wait_asynccnt(n)
#else
#define WAIT_ASYNC(n) asm volatile("s_wait_asynccnt %0" :: "i"(n) : "memory")
#endif
#endif

// ---------------------------------------------------------------------------
// Skinny MLP GEMM: Out(32 x N) = A1(32 x K1) @ W1^T (+ A2(32 x K2) @ W2^T)
//                  + bias, optional ReLU.  W row-major (N x K).
// One wave per 16-column tile; two 16-row M-tiles per wave; K step 32.
// BF16OUT writes __bf16 output (used for the ks tensor consumed by the conv).
// ---------------------------------------------------------------------------
template <bool BF16OUT>
__global__ __launch_bounds__(128) void mlp_kernel(
    const float* __restrict__ A1, const float* __restrict__ W1, int K1,
    const float* __restrict__ A2, const float* __restrict__ W2, int K2,
    const float* __restrict__ bias, void* __restrict__ OutP,
    int N, int do_relu)
{
    const int wave = threadIdx.x >> 5;
    const int lane = threadIdx.x & 31;
    const int nt   = blockIdx.x * (blockDim.x >> 5) + wave;
    if (nt * 16 >= N) return;                 // wave-uniform exit (EXEC stays full)
    const int n0 = nt * 16;
    const int l  = lane & 15;
    const int hi = lane >> 4;

    v8f acc0 = {}, acc1 = {};

    for (int kk = 0; kk < K1; kk += 32) {
        const int klo = kk + hi * 8;
        v16bf a0 = gld_cvt_a(A1 + (size_t)l * K1,        klo);
        v16bf a1 = gld_cvt_a(A1 + (size_t)(16 + l) * K1, klo);
        v16bf bf = gld_cvt_b(W1 + (size_t)(n0 + l) * K1 + kk + hi * 16);
        acc0 = wmma_bf16(a0, bf, acc0);
        acc1 = wmma_bf16(a1, bf, acc1);
    }
    if (W2) {
        for (int kk = 0; kk < K2; kk += 32) {
            const int klo = kk + hi * 8;
            v16bf a0 = gld_cvt_a(A2 + (size_t)l * K2,        klo);
            v16bf a1 = gld_cvt_a(A2 + (size_t)(16 + l) * K2, klo);
            v16bf bf = gld_cvt_b(W2 + (size_t)(n0 + l) * K2 + kk + hi * 16);
            acc0 = wmma_bf16(a0, bf, acc0);
            acc1 = wmma_bf16(a1, bf, acc1);
        }
    }

    const int   n  = n0 + l;
    const float bv = bias[n];
#pragma unroll
    for (int r = 0; r < 8; ++r) {
        const int m = r + hi * 8;
        float v0 = acc0[r] + bv;
        float v1 = acc1[r] + bv;
        if (do_relu) { v0 = v0 > 0.f ? v0 : 0.f; v1 = v1 > 0.f ? v1 : 0.f; }
        if (BF16OUT) {
            __bf16* Out = (__bf16*)OutP;
            Out[(size_t)m * N + n]        = (__bf16)v0;
            Out[(size_t)(m + 16) * N + n] = (__bf16)v1;
        } else {
            float* Out = (float*)OutP;
            Out[(size_t)m * N + n]        = v0;
            Out[(size_t)(m + 16) * N + n] = v1;
        }
    }
}

// ---------------------------------------------------------------------------
// Fused per-batch dynamic conv: h = relu(k_in @ x + b_in); out = k_out @ h + b_out
// k_in: 256x64, k_out: 64x256, x/out: 64x16384 per batch; h only ever in LDS.
// One workgroup (8 waves) per (batch, 128-pixel tile).
// ks arrives pre-converted to bf16 -> k_in/k_out staged via async LDS copies.
// ---------------------------------------------------------------------------
#define C_FH    256
#define C_FIN   64
#define C_FOUT  64
#define C_TN    128
#define LD_K    72      // sKin stride  [h][f]   (16B-aligned chunks, conflict-free)
#define LD_KO   264     // sKout stride [o][hh]
#define LD_X    72      // sXt stride   [s][f]
#define LD_H    264     // sHt stride   [s][hh]
#define KS_STR  33088

__global__ __launch_bounds__(256) void dynconv_kernel(
    const float* __restrict__ x, const __bf16* __restrict__ ks,
    float* __restrict__ out)
{
    __shared__ __align__(16) __bf16 sKin [C_FH * LD_K];    // 36.0 KB
    __shared__ __align__(16) __bf16 sKout[C_FOUT * LD_KO]; // 33.0 KB
    __shared__ __align__(16) __bf16 sXt  [C_TN * LD_X];    // 18.0 KB
    __shared__ __align__(16) __bf16 sHt  [C_TN * LD_H];    // 66.0 KB
    __shared__ float sBin[C_FH];
    __shared__ float sBout[C_FOUT];

    const int b    = blockIdx.y;
    const int s0   = blockIdx.x * C_TN;
    const int tid  = threadIdx.x;
    const int wave = tid >> 5;
    const int lane = tid & 31;
    const int l    = lane & 15;
    const int hi   = lane >> 4;

    const __bf16* kb = ks + (size_t)b * KS_STR;

    // ---- stage 0: stage k_in / k_out (bf16, 16B chunks) into padded LDS
#if USE_ASYNC_LDS
    for (int i = tid; i < 2048; i += 256) {            // k_in: 256 rows x 64
        const int row = i >> 3, c = (i & 7) << 3;
        async_copy16(kb + (i << 3), &sKin[row * LD_K + c]);
    }
    for (int i = tid; i < 2048; i += 256) {            // k_out: 64 rows x 256
        const int row = i >> 5, c = (i & 31) << 3;
        async_copy16(kb + 16384 + (i << 3), &sKout[row * LD_KO + c]);
    }
#else
    for (int i = tid; i < 2048; i += 256) {
        const int row = i >> 3, c = (i & 7) << 3;
        *(v8bf*)&sKin[row * LD_K + c] = *(const v8bf*)(kb + (i << 3));
    }
    for (int i = tid; i < 2048; i += 256) {
        const int row = i >> 5, c = (i & 31) << 3;
        *(v8bf*)&sKout[row * LD_KO + c] = *(const v8bf*)(kb + 16384 + (i << 3));
    }
#endif
    // biases (bf16 -> f32)
    for (int i = tid; i < C_FH; i += 256) sBin[i] = (float)kb[32768 + i];
    if (tid < C_FOUT)                     sBout[tid] = (float)kb[33024 + tid];

    // x tile, transposed [s][f], f32 -> bf16
    const float* xb = x + (((size_t)b * C_FIN) << 14) + s0;
    for (int i = tid; i < C_FIN * C_TN; i += 256) {
        const int f = i >> 7;
        const int j = i & 127;
        sXt[j * LD_X + f] = (__bf16)xb[((size_t)f << 14) + j];
    }
    // warm L2 for the next spatial tile of x (global_prefetch_b8)
    if (blockIdx.x + 1 < gridDim.x) {
        const int f = tid >> 2, c = (tid & 3) << 5;
        __builtin_prefetch(xb + C_TN + ((size_t)f << 14) + c, 0, 2);
    }

#if USE_ASYNC_LDS
    WAIT_ASYNC(8);          // k_in copies (first 8 async ops/wave) complete
#endif
    __syncthreads();

    // ---- stage 1: h(256x128) = relu(k_in @ xT + b_in) -> bf16 sHt[s][hh]
    {
        v16bf af[2][2];
#pragma unroll
        for (int t = 0; t < 2; ++t) {
            const __bf16* rp = &sKin[(wave * 32 + t * 16 + l) * LD_K];
            af[t][0] = lds_frag_a(rp,      hi * 8);
            af[t][1] = lds_frag_a(rp, 32 + hi * 8);
        }
#pragma unroll
        for (int nt = 0; nt < 8; ++nt) {
            const int n0 = nt * 16;
            const __bf16* cp = &sXt[(n0 + l) * LD_X];
            v16bf bf0 = lds_frag_b(cp + hi * 16);
            v16bf bf1 = lds_frag_b(cp + 32 + hi * 16);
#pragma unroll
            for (int t = 0; t < 2; ++t) {
                const int m0 = wave * 32 + t * 16;
                v8f acc = {};
                acc = wmma_bf16(af[t][0], bf0, acc);
                acc = wmma_bf16(af[t][1], bf1, acc);
                const int n = n0 + l;
                v8bf hv;
#pragma unroll
                for (int r = 0; r < 8; ++r) {
                    const int m = m0 + hi * 8 + r;
                    float v = acc[r] + sBin[m];
                    hv[r] = (__bf16)(v > 0.f ? v : 0.f);
                }
                *(v8bf*)&sHt[n * LD_H + m0 + hi * 8] = hv;
            }
        }
    }

#if USE_ASYNC_LDS
    WAIT_ASYNC(0);          // k_out copies complete (overlapped with stage 1)
#endif
    __syncthreads();

    // ---- stage 2: out(64x128) = k_out @ h + b_out  (K = 256)
    {
        const int m0  = (wave >> 1) * 16;
        const int ntb = (wave & 1) * 4;
        v16bf af[8];
        const __bf16* rp = &sKout[(m0 + l) * LD_KO];
#pragma unroll
        for (int k = 0; k < 8; ++k) af[k] = lds_frag_a(rp, k * 32 + hi * 8);
#pragma unroll
        for (int nt = ntb; nt < ntb + 4; ++nt) {
            const int n0 = nt * 16;
            const __bf16* cp = &sHt[(n0 + l) * LD_H];
            v8f acc = {};
#pragma unroll
            for (int k = 0; k < 8; ++k) {
                v16bf bf = lds_frag_b(cp + k * 32 + hi * 16);
                acc = wmma_bf16(af[k], bf, acc);
            }
            const int n = n0 + l;
            const size_t obase = (((size_t)b * C_FOUT) << 14) + s0 + n;
#pragma unroll
            for (int r = 0; r < 8; ++r) {
                const int o = m0 + hi * 8 + r;
                out[obase + ((size_t)o << 14)] = acc[r] + sBout[o];
            }
        }
    }
}

// ---------------------------------------------------------------------------
// Launch
// ---------------------------------------------------------------------------
extern "C" void kernel_launch(void* const* d_in, const int* in_sizes, int n_in,
                              void* d_out, int out_size, void* d_ws, size_t ws_size,
                              hipStream_t stream) {
    const float* x     = (const float*)d_in[0];   // (32, 64, 128, 128)
    const float* lat   = (const float*)d_in[1];   // (32, 512)
    const float* b1_w1 = (const float*)d_in[2];   // (512, 512)
    const float* b1_b1 = (const float*)d_in[3];   // (512,)
    const float* b1_w2 = (const float*)d_in[4];   // (512, 512)
    const float* b1_b2 = (const float*)d_in[5];   // (512,)
    const float* b1_ws = (const float*)d_in[6];   // (512, 512)
    const float* b2_w1 = (const float*)d_in[7];   // (1024, 512)
    const float* b2_b1 = (const float*)d_in[8];   // (1024,)
    const float* b2_w2 = (const float*)d_in[9];   // (33088, 1024)
    const float* b2_b2 = (const float*)d_in[10];  // (33088,)
    const float* b2_ws = (const float*)d_in[11];  // (33088, 512)

    float*  ws   = (float*)d_ws;
    float*  h1   = ws;                        // 32*512 f32
    float*  z    = ws + 16384;                // 32*512 f32
    float*  h2   = ws + 32768;                // 32*1024 f32
    __bf16* ksbf = (__bf16*)(ws + 65536);     // 32*33088 bf16 (16B aligned)

    // h1 = relu(lat @ b1_w1^T + b1_b1)
    mlp_kernel<false><<<dim3(8), 128, 0, stream>>>(lat, b1_w1, 512,
                                                   nullptr, nullptr, 0,
                                                   b1_b1, h1, 512, 1);
    // z = lat @ b1_ws^T + h1 @ b1_w2^T + b1_b2
    mlp_kernel<false><<<dim3(8), 128, 0, stream>>>(lat, b1_ws, 512,
                                                   h1, b1_w2, 512,
                                                   b1_b2, z, 512, 0);
    // h2 = relu(z @ b2_w1^T + b2_b1)
    mlp_kernel<false><<<dim3(16), 128, 0, stream>>>(z, b2_w1, 512,
                                                    nullptr, nullptr, 0,
                                                    b2_b1, h2, 1024, 1);
    // ks = z @ b2_ws^T + h2 @ b2_w2^T + b2_b2   -> bf16 (N = 33088 = 2068 tiles)
    mlp_kernel<true><<<dim3(517), 128, 0, stream>>>(z, b2_ws, 512,
                                                    h2, b2_w2, 1024,
                                                    b2_b2, ksbf, 33088, 0);
    // fused per-batch dynamic conv (h never leaves LDS)
    dynconv_kernel<<<dim3(128, 32), 256, 0, stream>>>(x, ksbf, (float*)d_out);
}
